// MambaBlock_49941879718189
// MI455X (gfx1250) — matrix operand order repaired and checked by reading, not compiled, not run on hardware
//
#include <hip/hip_runtime.h>
#include <hip/hip_bf16.h>

// ---------------------------------------------------------------------------
// Mamba block for MI455X (gfx1250, wave32).
//   GEMMs  -> v_wmma_f32_16x16x32_bf16 (fp32 accumulate), direct global b128
//             loads, same K-contiguous order (k = 16h + s) in A and B.
//   scan   -> register-resident state, 2 lanes per channel; B_t strips staged
//             into LDS by the Tensor Data Mover, double-buffered, synced with
//             s_wait_tensorcnt + workgroup barrier.
// ---------------------------------------------------------------------------

typedef __attribute__((ext_vector_type(16))) __bf16 v16bf;
typedef __attribute__((ext_vector_type(8)))  __bf16 v8bf;
typedef __attribute__((ext_vector_type(8)))  float  v8f;

#define BB   2
#define LL   1024
#define DM   1024
#define DI   2048
#define NS   64      // SSM state size
#define KC   4       // conv kernel
#define TCH  16      // scan timesteps per staged chunk
#define NCH  (LL / TCH)

// ---------------------------------------------------------------------------
// Tensor Data Mover: 1-row tile copy (nelem fp32, contiguous) into LDS.
// Descriptor fields per CDNA5 ISA.
// This toolchain exposes the 6-arg builtin:
//   (uint32x4 g0, int32x8 g1, int32x4 g2, int32x4 g3, int32x8 g?, i32 cpol)
// ---------------------------------------------------------------------------
#if __has_builtin(__builtin_amdgcn_tensor_load_to_lds)
#define HAVE_TDM 1
typedef __attribute__((ext_vector_type(4))) unsigned int v4u_t;
typedef __attribute__((ext_vector_type(8))) int          v8i_t;
typedef __attribute__((ext_vector_type(4))) int          v4i_t;

__device__ __forceinline__ void tdm_load_row(unsigned lds_addr, const void* gsrc,
                                             unsigned nelem) {
  const unsigned long long ga = (unsigned long long)gsrc;
  v4u_t g0;
  g0[0] = 1u;                                            // count=1 (valid), user mode
  g0[1] = lds_addr;                                      // [63:32]  lds byte address
  g0[2] = (unsigned)(ga & 0xFFFFFFFFu);                  // [95:64]  global addr lo
  g0[3] = (unsigned)((ga >> 32) & 0x01FFFFFFu)           // [120:96] global addr hi
        | (2u << 30);                                    // [127:126] type = 2 (image)
  v8i_t g1;
  g1[0] = (int)(2u << 16);            // wg_mask=0, data_size=2 (4 bytes), no flags
  g1[1] = (int)(nelem << 16);         // tensor_dim0[15:0]   at bits [63:48]
  g1[2] = (int)((nelem >> 16) | (1u << 16)); // tensor_dim0 hi | tensor_dim1=1
  g1[3] = (int)(nelem << 16);         // tensor_dim1 hi = 0 | tile_dim0 = nelem
  g1[4] = 1;                          // tile_dim1 = 1, tile_dim2 = 0
  g1[5] = (int)nelem;                 // tensor_dim0_stride lo
  g1[6] = 0;                          // stride hi / tensor_dim1_stride lo
  g1[7] = 0;
  const v4i_t z4 = {0, 0, 0, 0};      // groups 2/3 unused (<=2D tensor)
  const v8i_t z8 = {0, 0, 0, 0, 0, 0, 0, 0};
  __builtin_amdgcn_tensor_load_to_lds(g0, g1, z4, z4, z8, 0);
}
#endif

// ---------------------------------------------------------------------------
// fp32 -> bf16 elementwise convert
// ---------------------------------------------------------------------------
__global__ __launch_bounds__(256)
void cvt_bf16_kernel(const float* __restrict__ in, __bf16* __restrict__ out, int n) {
  int i = blockIdx.x * 256 + threadIdx.x;
  if (i < n) out[i] = (__bf16)in[i];
}

// ---------------------------------------------------------------------------
// RMSNorm over Dm, output bf16 (feeds the in_proj WMMA GEMM); block per row
// ---------------------------------------------------------------------------
__global__ __launch_bounds__(256)
void rmsnorm_kernel(const float* __restrict__ x, const float* __restrict__ w,
                    __bf16* __restrict__ xn) {
  __shared__ float red[256];
  const int row = blockIdx.x;
  const float* xr = x + (size_t)row * DM;
  float s = 0.f;
  for (int c = threadIdx.x; c < DM; c += 256) { float v = xr[c]; s += v * v; }
  red[threadIdx.x] = s;
  __syncthreads();
  for (int st = 128; st > 0; st >>= 1) {
    if (threadIdx.x < st) red[threadIdx.x] += red[threadIdx.x + st];
    __syncthreads();
  }
  const float inv = rsqrtf(red[0] * (1.0f / DM) + 1e-6f);
  __bf16* xo = xn + (size_t)row * DM;
  for (int c = threadIdx.x; c < DM; c += 256)
    xo[c] = (__bf16)(w[c] * xr[c] * inv);
}

// ---------------------------------------------------------------------------
// bf16 GEMM, C[M,N] = A[M,K] * B[N,K]^T, fp32 out.
//   block = 8 waves, block tile 128x128; wave tile 32x64 = 2x4 WMMA tiles.
//   A lane l: row m0+(l&15), halves K[hs*16..+7] ++ K[hs*16+8..+7]  (2x16B)
//   B lane l: col n0+(l&15), halves K[hs*16..+15]                  (1x32B)
//   C vgpr r: row m0+hs*8+r, col n0+(l&15)
// ---------------------------------------------------------------------------
__global__ __launch_bounds__(256)
void wmma_gemm_bf16nt(const __bf16* __restrict__ A, const __bf16* __restrict__ B,
                      float* __restrict__ Cm, int M, int N, int K) {
  const int lane = threadIdx.x & 31;
  const int wave = threadIdx.x >> 5;
  const int wm = wave & 3;
  const int wn = wave >> 2;
  const int m0 = blockIdx.x * 128 + wm * 32;
  const int n0 = blockIdx.y * 128 + wn * 64;
  if (m0 >= M || n0 >= N) return;          // wave-uniform: EXEC stays all-ones
  const int hs  = lane >> 4;
  const int l16 = lane & 15;

  v8f acc[2][4];
#pragma unroll
  for (int i = 0; i < 2; ++i)
#pragma unroll
    for (int j = 0; j < 4; ++j)
#pragma unroll
      for (int r = 0; r < 8; ++r) acc[i][j][r] = 0.0f;

  const __bf16* arow[2];
  arow[0] = A + (size_t)(m0 + l16) * K;
  arow[1] = A + (size_t)(m0 + 16 + l16) * K;
  const __bf16* brow[4];
#pragma unroll
  for (int j = 0; j < 4; ++j) brow[j] = B + (size_t)(n0 + j * 16 + l16) * K;

  for (int k0 = 0; k0 < K; k0 += 32) {
    // latency hedge for the first-touch streaming pass (global_prefetch_b8)
    if (k0 + 64 < K) {
      __builtin_prefetch((const void*)(arow[0] + k0 + 64), 0, 0);
      __builtin_prefetch((const void*)(brow[0] + k0 + 64), 0, 0);
      __builtin_prefetch((const void*)(brow[2] + k0 + 64), 0, 0);
    }
    v16bf a[2];
#pragma unroll
    for (int i = 0; i < 2; ++i) {
      const __bf16* p = arow[i] + k0 + hs * 16;
      v8bf lo = *(const v8bf*)p;
      v8bf hi = *(const v8bf*)(p + 8);
#pragma unroll
      for (int t = 0; t < 8; ++t) { a[i][t] = lo[t]; a[i][8 + t] = hi[t]; }
    }
    v16bf bfrag[4];
#pragma unroll
    for (int j = 0; j < 4; ++j)
      bfrag[j] = *(const v16bf*)(brow[j] + k0 + hs * 16);

#pragma unroll
    for (int i = 0; i < 2; ++i)
#pragma unroll
      for (int j = 0; j < 4; ++j)
        acc[i][j] = __builtin_amdgcn_wmma_f32_16x16x32_bf16(
            false, a[i], false, bfrag[j], (short)0, acc[i][j], false, false);
  }

#pragma unroll
  for (int i = 0; i < 2; ++i) {
    const int rbase = m0 + i * 16 + hs * 8;
#pragma unroll
    for (int j = 0; j < 4; ++j) {
      const int col = n0 + j * 16 + l16;
#pragma unroll
      for (int r = 0; r < 8; ++r)
        Cm[(size_t)(rbase + r) * N + col] = acc[i][j][r];
    }
  }
}

// ---------------------------------------------------------------------------
// causal depthwise conv1d (K=4) + bias + SiLU. Reads xi = xr[:, 0:DI].
// ---------------------------------------------------------------------------
__global__ __launch_bounds__(256)
void conv_silu_kernel(const float* __restrict__ xr, const float* __restrict__ cw,
                      const float* __restrict__ cb, float* __restrict__ xc,
                      __bf16* __restrict__ xcb) {
  const int idx = blockIdx.x * 256 + threadIdx.x;   // over B*L*DI
  const int d = idx & (DI - 1);
  const int l = (idx >> 11) & (LL - 1);
  const int b = idx >> 21;
  float s = cb[d];
#pragma unroll
  for (int k = 0; k < KC; ++k) {
    const int li = l - (KC - 1) + k;
    if (li >= 0)
      s += cw[d * KC + k] * xr[(size_t)(b * LL + li) * (2 * DI) + d];
  }
  const float v = s / (1.0f + __expf(-s));
  xc[idx]  = v;
  xcb[idx] = (__bf16)v;
}

// ---------------------------------------------------------------------------
// dt = softplus(dt_raw + dt_b), in place
// ---------------------------------------------------------------------------
__global__ __launch_bounds__(256)
void softplus_bias_kernel(float* __restrict__ dt, const float* __restrict__ dtb) {
  const int idx = blockIdx.x * 256 + threadIdx.x;
  const int d = idx & (DI - 1);
  const float v = dt[idx] + dtb[d];
  dt[idx] = (v > 20.0f) ? v : __logf(1.0f + __expf(v));
}

// ---------------------------------------------------------------------------
// selective scan: h[n] = exp(A[d,n]*dt)*h[n] + B_t[n]*x_t ; y = sum_n h + D*x
// one (b,d) channel per LANE PAIR (32 states each, in VGPRs).
// B_t strips (TCH x NS fp32 = 4KB, contiguous) staged into LDS by the TDM,
// double-buffered two chunks ahead; wave 0 issues + waits TENSORcnt, the
// workgroup barrier publishes the buffer to the other 7 waves.
// grid = 32 blocks x 256 threads = 8192 lanes = 2 * B * DI
// ---------------------------------------------------------------------------
__global__ __launch_bounds__(256)
void scan_kernel(const float* __restrict__ xc, const float* __restrict__ dt,
                 const float* __restrict__ Bs, const float* __restrict__ A_log,
                 const float* __restrict__ Dp, float* __restrict__ y) {
  __shared__ float bsh[2][TCH * NS];           // 2 x 4 KB double buffer
  const int tid  = threadIdx.x;
  const int half = tid & 1;                    // which 32 of the 64 states
  const int p    = tid >> 1;                   // channel within block (0..127)
  const int b    = blockIdx.x >> 4;
  const int d    = (blockIdx.x & 15) * 128 + p;

  float Areg[32], h[32];
  const float* al = A_log + (size_t)d * NS + half * 32;
#pragma unroll
  for (int n = 0; n < 32; ++n) { Areg[n] = -__expf(al[n]); h[n] = 0.0f; }
  const float Dv = Dp[d];
  const size_t rowbase = (size_t)b * LL;
  const float* bs_base = Bs + rowbase * NS;

#if defined(HAVE_TDM)
  const unsigned lds0 = (unsigned)(size_t)(&bsh[0][0]);
  const unsigned lds1 = (unsigned)(size_t)(&bsh[1][0]);
  if (tid < 32) {                              // wave 0 drives the TDM
    tdm_load_row(lds0, bs_base,            TCH * NS);
    tdm_load_row(lds1, bs_base + TCH * NS, TCH * NS);
  }
  for (int c = 0; c < NCH; ++c) {
    if (tid < 32) {
      // TENSORcnt is per-wave and in-order: <=1 leaves only the prefetched
      // next chunk in flight; the final chunk needs a full drain.
      if (c + 1 < NCH) __builtin_amdgcn_s_wait_tensorcnt(1);
      else             __builtin_amdgcn_s_wait_tensorcnt(0);
    }
    __syncthreads();                           // chunk c visible to all waves
    const float* bchunk = &bsh[c & 1][0];
#pragma unroll 4
    for (int tt = 0; tt < TCH; ++tt) {
      const size_t idx = (rowbase + c * TCH + tt) * DI + d;
      const float xv  = xc[idx];
      const float dtv = dt[idx];
      const float* bp = bchunk + tt * NS + half * 32;
      float sum = 0.0f;
#pragma unroll
      for (int n = 0; n < 32; ++n) {
        h[n] = __expf(Areg[n] * dtv) * h[n] + bp[n] * xv;
        sum += h[n];
      }
      sum += __shfl_xor(sum, 1, 32);           // combine the two state halves
      if (half == 0) y[idx] = sum + Dv * xv;
    }
    __syncthreads();                           // all waves done with buf[c&1]
    if (tid < 32 && c + 2 < NCH)
      tdm_load_row((c & 1) ? lds1 : lds0, bs_base + (size_t)(c + 2) * TCH * NS,
                   TCH * NS);
  }
#else
  for (int t0 = 0; t0 < LL; t0 += TCH) {
    __syncthreads();
    const float* src = bs_base + (size_t)t0 * NS;
    for (int e = tid; e < TCH * NS; e += 256) bsh[0][e] = src[e];
    __syncthreads();
#pragma unroll 4
    for (int tt = 0; tt < TCH; ++tt) {
      const size_t idx = (rowbase + t0 + tt) * DI + d;
      const float xv  = xc[idx];
      const float dtv = dt[idx];
      const float* bp = &bsh[0][tt * NS + half * 32];
      float sum = 0.0f;
#pragma unroll
      for (int n = 0; n < 32; ++n) {
        h[n] = __expf(Areg[n] * dtv) * h[n] + bp[n] * xv;
        sum += h[n];
      }
      sum += __shfl_xor(sum, 1, 32);
      if (half == 0) y[idx] = sum + Dv * xv;
    }
  }
#endif
}

// ---------------------------------------------------------------------------
// ymul = y * silu(res),  res = xr[:, DI:2*DI]; bf16 out for out_proj GEMM
// ---------------------------------------------------------------------------
__global__ __launch_bounds__(256)
void gate_kernel(const float* __restrict__ y, const float* __restrict__ xr,
                 __bf16* __restrict__ ymb) {
  const int idx = blockIdx.x * 256 + threadIdx.x;  // over (B*L)*DI
  const int row = idx >> 11;
  const int d   = idx & (DI - 1);
  const float r = xr[(size_t)row * (2 * DI) + DI + d];
  const float s = r / (1.0f + __expf(-r));
  ymb[idx] = (__bf16)(y[idx] * s);
}

// ---------------------------------------------------------------------------
// out = gemm4_out + skip(x)
// ---------------------------------------------------------------------------
__global__ __launch_bounds__(256)
void add_skip_kernel(const float* __restrict__ g4, const float* __restrict__ x,
                     float* __restrict__ out) {
  const int idx = blockIdx.x * 256 + threadIdx.x;
  out[idx] = g4[idx] + x[idx];
}

// ---------------------------------------------------------------------------
extern "C" void kernel_launch(void* const* d_in, const int* in_sizes, int n_in,
                              void* d_out, int out_size, void* d_ws, size_t ws_size,
                              hipStream_t stream) {
  (void)in_sizes; (void)n_in; (void)out_size; (void)ws_size;
  const float* x        = (const float*)d_in[0];
  const float* norm_w   = (const float*)d_in[1];
  const float* in_w     = (const float*)d_in[2];
  const float* conv_w   = (const float*)d_in[3];
  const float* conv_b   = (const float*)d_in[4];
  const float* xproj_w  = (const float*)d_in[5];
  const float* dt_w     = (const float*)d_in[6];
  const float* dt_b     = (const float*)d_in[7];
  const float* A_log    = (const float*)d_in[8];
  const float* D_param  = (const float*)d_in[9];
  const float* out_w    = (const float*)d_in[10];
  float* out = (float*)d_out;

  const int M  = BB * LL;          // 2048 rows
  char* wsp = (char*)d_ws;
  size_t off = 0;
  auto carve = [&](size_t bytes) -> void* {
    void* p = wsp + off;
    off = (off + bytes + 255) & ~(size_t)255;
    return p;
  };
  __bf16* xn_bf  = (__bf16*)carve((size_t)M * DM * 2);
  __bf16* inw_bf = (__bf16*)carve((size_t)2 * DI * DM * 2);
  __bf16* xpw_bf = (__bf16*)carve((size_t)2 * NS * DI * 2);
  __bf16* dtw_bf = (__bf16*)carve((size_t)DI * DI * 2);
  __bf16* ow_bf  = (__bf16*)carve((size_t)DM * DI * 2);
  float*  xr     = (float*) carve((size_t)M * 2 * DI * 4);
  float*  xc     = (float*) carve((size_t)M * DI * 4);
  __bf16* xc_bf  = (__bf16*)carve((size_t)M * DI * 2);
  float*  dtbuf  = (float*) carve((size_t)M * DI * 4);
  float*  bssm   = (float*) carve((size_t)M * NS * 4);
  float*  ybuf   = (float*) carve((size_t)M * DI * 4);
  __bf16* ym_bf  = (__bf16*)carve((size_t)M * DI * 2);
  float*  g4     = (float*) carve((size_t)M * DM * 4);

  // weight casts (bf16 feeds the matrix cores)
  cvt_bf16_kernel<<<(2 * DI * DM) / 256, 256, 0, stream>>>(in_w,    inw_bf, 2 * DI * DM);
  cvt_bf16_kernel<<<(2 * NS * DI) / 256, 256, 0, stream>>>(xproj_w, xpw_bf, 2 * NS * DI);
  cvt_bf16_kernel<<<(DI * DI)     / 256, 256, 0, stream>>>(dt_w,    dtw_bf, DI * DI);
  cvt_bf16_kernel<<<(DM * DI)     / 256, 256, 0, stream>>>(out_w,   ow_bf,  DM * DI);

  // 1) RMSNorm -> bf16
  rmsnorm_kernel<<<M, 256, 0, stream>>>(x, norm_w, xn_bf);

  // 2) in_proj: [2048,1024] x [4096,1024]^T -> xr [2048,4096]
  wmma_gemm_bf16nt<<<dim3(M / 128, (2 * DI) / 128), 256, 0, stream>>>(
      xn_bf, inw_bf, xr, M, 2 * DI, DM);

  // 3) causal depthwise conv + SiLU
  conv_silu_kernel<<<(M * DI) / 256, 256, 0, stream>>>(xr, conv_w, conv_b, xc, xc_bf);

  // 4) B_ssm only (delta half of x_proj is unused in the reference):
  //    [2048,2048] x [64,2048]^T -> bssm [2048,64]
  wmma_gemm_bf16nt<<<dim3(M / 128, 1), 256, 0, stream>>>(
      xc_bf, xpw_bf + (size_t)NS * DI, bssm, M, NS, DI);

  // 5) dt GEMM: [2048,2048] x [2048,2048]^T -> dtbuf, then softplus(+bias)
  wmma_gemm_bf16nt<<<dim3(M / 128, DI / 128), 256, 0, stream>>>(
      xc_bf, dtw_bf, dtbuf, M, DI, DI);
  softplus_bias_kernel<<<(M * DI) / 256, 256, 0, stream>>>(dtbuf, dt_b);

  // 6) selective scan (recurrence over L), TDM-staged B_t
  scan_kernel<<<(2 * BB * DI) / 256, 256, 0, stream>>>(xc, dtbuf, bssm, A_log,
                                                       D_param, ybuf);

  // 7) gate: y * silu(res) -> bf16
  gate_kernel<<<(M * DI) / 256, 256, 0, stream>>>(ybuf, xr, ym_bf);

  // 8) out_proj: [2048,2048] x [1024,2048]^T -> g4, then + skip
  wmma_gemm_bf16nt<<<dim3(M / 128, DM / 128), 256, 0, stream>>>(
      ym_bf, ow_bf, g4, M, DM, DI);
  add_skip_kernel<<<(M * DM) / 256, 256, 0, stream>>>(g4, x, out);
}
